// MassivePool_44066364457356
// MI455X (gfx1250) — matrix operand order repaired and checked by reading, not compiled, not run on hardware
//
#include <hip/hip_runtime.h>

// ---------------------------------------------------------------------------
// Problem constants (from the reference): B=2,S=1024 -> T=2048 tokens
// ---------------------------------------------------------------------------
#define RDIM   512        // query/key feature dim
#define NKEYS  100000     // pool size
#define PDIM   1024       // pool/value dim
#define TOKENS 2048       // B*S
#define TOPK   32

typedef __bf16 bf16_t;
typedef __attribute__((ext_vector_type(16))) __bf16 v16bf;
typedef __attribute__((ext_vector_type(8)))  __bf16 v8bf;
typedef __attribute__((ext_vector_type(4)))  __bf16 v4bf;
typedef __attribute__((ext_vector_type(8)))  float  v8f;

#if __has_builtin(__builtin_amdgcn_sched_barrier)
#define SCHED_FENCE() __builtin_amdgcn_sched_barrier(0)
#else
#define SCHED_FENCE()
#endif

// ---------------------------------------------------------------------------
// f32 -> bf16 bulk convert (sizes are exact multiples of 1024 elements)
// ---------------------------------------------------------------------------
__global__ __launch_bounds__(256) void k_cvt_bf16(const float* __restrict__ src,
                                                  bf16_t* __restrict__ dst) {
    size_t i = ((size_t)blockIdx.x * blockDim.x + threadIdx.x) * 4;
    float4 v = *(const float4*)(src + i);
    v4bf o = { (bf16_t)v.x, (bf16_t)v.y, (bf16_t)v.z, (bf16_t)v.w };
    *(v4bf*)(dst + i) = o;
}

// ---------------------------------------------------------------------------
// Kernel 1: scores GEMM (bf16 WMMA) fused with streaming top-32 + softmax.
// Block = 16 tokens x 128 key-columns per chunk (8 waves, one 16x16 tile each).
// B fragments flow through a depth-4 register pipeline that is pinned with
// sched_barrier(0) so loads stay 4 stages ahead of their consuming WMMA; the
// stream is flat ACROSS chunks (kc=12..15 prefetch the next chunk's stages),
// so ~8 b128 loads remain in flight through the per-chunk barrier.
// Candidate queue is ping-ponged (2 banks) -> exactly 1 barrier per chunk,
// and the serial merge (wave 0) overlaps the next chunk's WMMA phase.
// ---------------------------------------------------------------------------
#define CAND_MAX 2048
#define KCHUNKS  (RDIM / 32)   // 16

__global__ __launch_bounds__(256) void k_scores_topk(
    const bf16_t* __restrict__ qb,   // [TOKENS][RDIM] bf16
    const bf16_t* __restrict__ kb,   // [NKEYS][RDIM]  bf16
    int*   __restrict__ topi_g,      // [TOKENS][TOPK]
    float* __restrict__ topw_g)      // [TOKENS][TOPK] softmax weights
{
    __shared__ float    s_cscore[2][CAND_MAX];
    __shared__ unsigned s_cmeta [2][CAND_MAX];
    __shared__ float    s_tops[16][TOPK];
    __shared__ int      s_topi[16][TOPK];
    __shared__ float    s_min [16];       // monotone non-decreasing thresholds
    __shared__ int      s_mpos[16];
    __shared__ int      s_cnt [2];

    const int tid  = threadIdx.x;
    const int lane = tid & 31;
    const int wave = tid >> 5;
    const int half = lane >> 4;          // 0: lanes 0-15, 1: lanes 16-31
    const int m0   = blockIdx.x * 16;    // token-tile base
    const int mrow = m0 + (lane & 15);

    if (tid < 16) {
        s_min[tid] = -3.4e38f; s_mpos[tid] = 0;
        for (int j = 0; j < TOPK; ++j) { s_tops[tid][j] = -3.4e38f; s_topi[tid][j] = 0; }
    }
    if (tid == 0) { s_cnt[0] = 0; s_cnt[1] = 0; }

    // Preload A fragments for all 16 K-chunks (contiguous K order, same as B at line 96):
    // lane<16: elems 0..15 = K 0..15 (half=1 shifts by 16)
    v16bf afrag[KCHUNKS];
    #pragma unroll
    for (int kc = 0; kc < KCHUNKS; ++kc) {
        const bf16_t* q = qb + (size_t)mrow * RDIM + kc * 32 + half * 16;
        v8bf lo = *(const v8bf*)(q);
        v8bf hi = *(const v8bf*)(q + 8);
        afrag[kc] = __builtin_shufflevector(lo, hi, 0,1,2,3,4,5,6,7,8,9,10,11,12,13,14,15);
    }
    __syncthreads();

    // Per-wave B-column base pointer for a given chunk (tail columns clamped;
    // invalid columns are filtered at candidate-push time).
    auto colptr = [&](int blk_) -> const bf16_t* {
        int col_  = blk_ * 128 + wave * 16 + (lane & 15);
        int colc_ = col_ < NKEYS ? col_ : NKEYS - 1;
        return kb + (size_t)colc_ * RDIM + half * 16;
    };
    auto loadBat = [&](const bf16_t* base, int kc) -> v16bf {
        // B 32x16 layout: lane holds 16 contiguous K values of its column
        return *(const v16bf*)(base + kc * 32);
    };

    const int nChunks = (NKEYS + 127) / 128;

    // Warm the pipeline with chunk 0's first 4 stages.
    const bf16_t* krow = colptr(0);
    v16bf bb[4];
    #pragma unroll
    for (int i = 0; i < 4; ++i) bb[i] = loadBat(krow, i);

    for (int blk = 0; blk < nChunks; ++blk) {
        const int p = blk & 1;
        // Prepare the other bank for the NEXT chunk's pushes. That bank was
        // fully consumed by this wave's merge in the previous chunk (program order).
        if (tid == 0) s_cnt[1 - p] = 0;

        const int col = blk * 128 + wave * 16 + (lane & 15);
        const bf16_t* krow_nxt = colptr(blk + 1 < nChunks ? blk + 1 : blk);

        v8f c = {0.f, 0.f, 0.f, 0.f, 0.f, 0.f, 0.f, 0.f};
        #pragma unroll
        for (int kc = 0; kc < KCHUNKS; ++kc) {
            v16bf cur = bb[kc & 3];
            // Refill 4 stages ahead; tail iterations pre-warm the next chunk.
            if (kc + 4 < KCHUNKS) bb[kc & 3] = loadBat(krow,     kc + 4);
            else                  bb[kc & 3] = loadBat(krow_nxt, kc + 4 - KCHUNKS);
            SCHED_FENCE();   // pin: loads issue BEFORE this wmma
            c = __builtin_amdgcn_wmma_f32_16x16x32_bf16(
                    false, afrag[kc], false, cur, (short)0, c, false, false);
            SCHED_FENCE();
        }
        krow = krow_nxt;

        // Register-level threshold filter -> candidate bank p.
        // Concurrent s_min reads during wave0's previous merge are safe:
        // a stale (lower) threshold only admits extra candidates.
        const int rbase = half * 8;
        #pragma unroll
        for (int j = 0; j < 8; ++j) {
            float s = c[j];
            int row = rbase + j;
            if (col < NKEYS && s > s_min[row]) {
                int pos = atomicAdd(&s_cnt[p], 1);
                if (pos < CAND_MAX) {
                    s_cscore[p][pos] = s;
                    s_cmeta [p][pos] = ((unsigned)col << 4) | (unsigned)row;
                }
            }
        }
        __syncthreads();   // the ONLY barrier per chunk: bank-p pushes visible

        // Wave 0 merges bank p; waves 1-7 run ahead into chunk blk+1 (they
        // only touch bank 1-p). No barrier inside -> divergence-free skip.
        if (tid < 16) {
            int cnt = s_cnt[p]; if (cnt > CAND_MAX) cnt = CAND_MAX;
            if (cnt) {
                float mn = s_min[tid]; int mp = s_mpos[tid];
                for (int i = 0; i < cnt; ++i) {
                    unsigned meta = s_cmeta[p][i];
                    if ((int)(meta & 15u) != tid) continue;
                    float s = s_cscore[p][i];
                    if (s <= mn) continue;
                    s_tops[tid][mp] = s;
                    s_topi[tid][mp] = (int)(meta >> 4);
                    mn = s_tops[tid][0]; mp = 0;
                    for (int j = 1; j < TOPK; ++j) {
                        float v = s_tops[tid][j];
                        if (v < mn) { mn = v; mp = j; }
                    }
                }
                s_min[tid] = mn; s_mpos[tid] = mp;
            }
        }
    }

    // Softmax over the 32 winners; done by the same wave that merged, so no
    // extra barrier is required. Order-free: downstream is a weighted sum.
    if (tid < 16) {
        const int t = m0 + tid;
        float mx = -3.4e38f;
        for (int j = 0; j < TOPK; ++j) mx = fmaxf(mx, s_tops[tid][j]);
        float sum = 0.f;
        for (int j = 0; j < TOPK; ++j) {
            float e = expf(s_tops[tid][j] - mx);
            s_tops[tid][j] = e; sum += e;
        }
        float inv = 1.f / sum;
        for (int j = 0; j < TOPK; ++j) {
            topw_g[t * TOPK + j] = s_tops[tid][j] * inv;
            topi_g[t * TOPK + j] = s_topi[tid][j];
        }
    }
}

// ---------------------------------------------------------------------------
// Kernel 2: gather + weighted sum over top-32 pool rows. HBM-bound (~268 MB).
// One block per token; thread owns 4 consecutive P-columns. Output as bf16
// A-operand for the projection GEMM.
// ---------------------------------------------------------------------------
__global__ __launch_bounds__(256) void k_aggregate(
    const float* __restrict__ pool,     // [NKEYS][PDIM] f32
    const int*   __restrict__ topi_g,
    const float* __restrict__ topw_g,
    bf16_t* __restrict__ aggb)          // [TOKENS][PDIM] bf16
{
    __shared__ int   sidx[TOPK];
    __shared__ float sw  [TOPK];
    const int t   = blockIdx.x;
    const int tid = threadIdx.x;
    if (tid < TOPK) {
        sidx[tid] = topi_g[t * TOPK + tid];
        sw  [tid] = topw_g[t * TOPK + tid];
    }
    __syncthreads();

    const int col = tid * 4;
    float ax = 0.f, ay = 0.f, az = 0.f, aw = 0.f;
    #pragma unroll 8
    for (int k = 0; k < TOPK; ++k) {
        const float4 p = *(const float4*)(pool + (size_t)sidx[k] * PDIM + col);
        float w = sw[k];
        ax += w * p.x; ay += w * p.y; az += w * p.z; aw += w * p.w;
    }
    v4bf o = { (bf16_t)ax, (bf16_t)ay, (bf16_t)az, (bf16_t)aw };
    *(v4bf*)(aggb + (size_t)t * PDIM + col) = o;
}

// ---------------------------------------------------------------------------
// Kernel 3: output projection out = agg @ w_out^T via bf16 WMMA.
// Block = 16 rows x 128 cols (8 waves x one 16x16 tile), K = 1024.
// A and B fragments are software-pipelined 2 deep, schedule pinned.
// ---------------------------------------------------------------------------
__global__ __launch_bounds__(256) void k_outproj(
    const bf16_t* __restrict__ aggb,   // [TOKENS][PDIM] bf16
    const bf16_t* __restrict__ wb,     // [PDIM][PDIM]   bf16 (w_out row-major)
    float* __restrict__ out)           // [TOKENS][PDIM] f32
{
    const int tid  = threadIdx.x;
    const int lane = tid & 31;
    const int wave = tid >> 5;
    const int half = lane >> 4;
    const int m0   = blockIdx.x * 16;
    const int mrow = m0 + (lane & 15);
    const int q0   = blockIdx.y * 128 + wave * 16;
    const int qcol = q0 + (lane & 15);

    auto loadA = [&](int kc) -> v16bf {
        const bf16_t* a = aggb + (size_t)mrow * PDIM + kc * 32 + half * 16;
        v8bf lo = *(const v8bf*)(a);
        v8bf hi = *(const v8bf*)(a + 8);
        return __builtin_shufflevector(lo, hi, 0,1,2,3,4,5,6,7,8,9,10,11,12,13,14,15);
    };
    auto loadB = [&](int kc) -> v16bf {
        // B column q = w_out row q, contiguous along K=p
        return *(const v16bf*)(wb + (size_t)qcol * PDIM + kc * 32 + half * 16);
    };

    v16bf aP[2] = { loadA(0), loadA(1) };
    v16bf bP[2] = { loadB(0), loadB(1) };
    v8f c = {0.f, 0.f, 0.f, 0.f, 0.f, 0.f, 0.f, 0.f};
    #pragma unroll
    for (int kc = 0; kc < PDIM / 32; ++kc) {
        v16bf ac = aP[kc & 1];
        v16bf bc = bP[kc & 1];
        if (kc + 2 < PDIM / 32) {
            aP[kc & 1] = loadA(kc + 2);
            bP[kc & 1] = loadB(kc + 2);
        }
        SCHED_FENCE();   // pin: next-stage loads issue before this wmma
        c = __builtin_amdgcn_wmma_f32_16x16x32_bf16(
                false, ac, false, bc, (short)0, c, false, false);
        SCHED_FENCE();
    }
    const int rbase = half * 8;
    #pragma unroll
    for (int j = 0; j < 8; ++j)
        out[(size_t)(m0 + rbase + j) * PDIM + qcol] = c[j];
}

// ---------------------------------------------------------------------------
// Launch: inputs are {query, keys, pool, w_out} (f32). Output f32 [2,1024,1024].
// Workspace layout (256B-aligned, ~111.3 MB total):
//   kb 102,400,000 | qb 2,097,152 | wb 2,097,152 | aggb 4,194,304 |
//   topi 262,144   | topw 262,144
// ---------------------------------------------------------------------------
extern "C" void kernel_launch(void* const* d_in, const int* in_sizes, int n_in,
                              void* d_out, int out_size, void* d_ws, size_t ws_size,
                              hipStream_t stream) {
    const float* query = (const float*)d_in[0];   // [2,1024,512]
    const float* keys  = (const float*)d_in[1];   // [100000,512]
    const float* pool  = (const float*)d_in[2];   // [100000,1024]
    const float* w_out = (const float*)d_in[3];   // [1024,1024]
    float* out = (float*)d_out;

    char* ws = (char*)d_ws;
    bf16_t* kb   = (bf16_t*)(ws);
    bf16_t* qb   = (bf16_t*)(ws + 102400000u);
    bf16_t* wb   = (bf16_t*)(ws + 104497152u);
    bf16_t* aggb = (bf16_t*)(ws + 106594304u);
    int*    topi = (int*)   (ws + 110788608u);
    float*  topw = (float*) (ws + 111050752u);

    // 1) precision conversion (one-shot, HBM-bound, ~10us)
    k_cvt_bf16<<<50000, 256, 0, stream>>>(keys,  kb);  // 51.2M elems
    k_cvt_bf16<<<1024,  256, 0, stream>>>(query, qb);  // 1.05M elems
    k_cvt_bf16<<<1024,  256, 0, stream>>>(w_out, wb);  // 1.05M elems

    // 2) scores GEMM + fused streaming top-32 + softmax
    k_scores_topk<<<TOKENS / 16, 256, 0, stream>>>(qb, kb, topi, topw);

    // 3) gather + weighted aggregation (-> bf16)
    k_aggregate<<<TOKENS, 256, 0, stream>>>(pool, topi, topw, aggb);

    // 4) output projection GEMM
    k_outproj<<<dim3(TOKENS / 16, PDIM / 128), 256, 0, stream>>>(aggb, wb, out);
}
